// GraphAttnLayer_70291434766740
// MI455X (gfx1250) — matrix operand order repaired and checked by reading, not compiled, not run on hardware
//
#include <hip/hip_runtime.h>

#define NN 50000
#define EE 800000
#define TT 2
#define INF_ 128
#define HH 4
#define DHD 32
#define OUTF 128
#define NEG_SLOPE 0.2f
#define EPS_LN 1e-12f

typedef __attribute__((ext_vector_type(16))) _Float16 v16h;
typedef __attribute__((ext_vector_type(8)))  _Float16 v8h;
typedef __attribute__((ext_vector_type(8)))  float    v8f;

// ---------------- order-preserving float <-> uint for atomic segment-max ----
__device__ __forceinline__ unsigned f2ord(float f) {
    unsigned u = __float_as_uint(f);
    return (u & 0x80000000u) ? ~u : (u | 0x80000000u);
}
__device__ __forceinline__ float ord2f(unsigned u) {
    unsigned v = (u & 0x80000000u) ? (u & 0x7FFFFFFFu) : ~u;
    return __uint_as_float(v);
}

// ---------------- 0) transpose W -> f16, layout [T][OUT][IN] ---------------
__global__ void wtrans_kernel(const float* __restrict__ W, _Float16* __restrict__ WT)
{
    int i = blockIdx.x * blockDim.x + threadIdx.x;      // indexes W[t][k][o]
    if (i >= TT * INF_ * OUTF) return;
    int o = i % OUTF;
    int k = (i / OUTF) % INF_;
    int t = i / (OUTF * INF_);
    WT[((size_t)t * OUTF + o) * INF_ + k] = (_Float16)W[i];
}

// ---------------- 1) projection: fh[t] = feature @ W[t] via v_wmma ----------
// One wave computes a 16x128 stripe: A loaded once (vectorized float4) and
// reused across 8 column tiles; B from pre-transposed f16 W via b128 loads.
__global__ __launch_bounds__(32) void proj_wmma_kernel(
    const float* __restrict__ feature, const _Float16* __restrict__ WT,
    float* __restrict__ fh)
{
    const int nTiles = NN / 16;              // 3125 row tiles
    const int bid  = blockIdx.x;             // TT * 3125 blocks
    const int t    = bid / nTiles;
    const int rt   = bid % nTiles;
    const int lane = threadIdx.x;
    const int row0 = rt * 16;

    float* fht = fh + (size_t)t * NN * OUTF;

    const int Mloc = lane & 15;
    const int hi   = lane >> 4;              // 0: lanes 0-15, 1: lanes 16-31
    const float* Arow = feature + (size_t)(row0 + Mloc) * INF_;

    // warm L2 for the next row tile (global_prefetch_b8)
    if (rt + 1 < nTiles) __builtin_prefetch(Arow + 16 * INF_, 0, 1);

    // A-matrix per-lane layout (16-bit A 16x32, ISA 7.12.2):
    //  halves 0-7  -> K = kc*32 + hi*8      .. +7   (contiguous run of 8)
    //  halves 8-15 -> K = kc*32 + 16 + hi*8 .. +7   (contiguous run of 8)
    v16h a[4];
#pragma unroll
    for (int kc = 0; kc < 4; ++kc) {
        const float4* A4 = (const float4*)(Arow + kc * 32 + hi * 8);
        float4 p0 = A4[0], p1 = A4[1];                 // K run 0
        const float4* A4b = (const float4*)(Arow + kc * 32 + 16 + hi * 8);
        float4 p2 = A4b[0], p3 = A4b[1];               // K run 1
        a[kc][0]  = (_Float16)p0.x; a[kc][1]  = (_Float16)p0.y;
        a[kc][2]  = (_Float16)p0.z; a[kc][3]  = (_Float16)p0.w;
        a[kc][4]  = (_Float16)p1.x; a[kc][5]  = (_Float16)p1.y;
        a[kc][6]  = (_Float16)p1.z; a[kc][7]  = (_Float16)p1.w;
        a[kc][8]  = (_Float16)p2.x; a[kc][9]  = (_Float16)p2.y;
        a[kc][10] = (_Float16)p2.z; a[kc][11] = (_Float16)p2.w;
        a[kc][12] = (_Float16)p3.x; a[kc][13] = (_Float16)p3.y;
        a[kc][14] = (_Float16)p3.z; a[kc][15] = (_Float16)p3.w;
    }

    const int Nloc = lane & 15;
    const int Kb   = hi * 8;                 // B: lane<16 -> K 0..7,16..23, else 8..15,24..31
#pragma unroll
    for (int ct = 0; ct < 8; ++ct) {
        // transposed-W row for this lane's output column
        const _Float16* Wrow = WT + ((size_t)t * OUTF + ct * 16 + Nloc) * INF_;
        v8f acc = {0.f, 0.f, 0.f, 0.f, 0.f, 0.f, 0.f, 0.f};
#pragma unroll
        for (int kc = 0; kc < 4; ++kc) {
            const v8h* B8 = (const v8h*)(Wrow + kc * 32 + Kb);  // two runs of 8 halves, 16 apart
            v8h blo = B8[0], bhi = B8[2];
            v16h b;
#pragma unroll
            for (int q = 0; q < 8; ++q) { b[q] = blo[q]; b[8 + q] = bhi[q]; }
            acc = __builtin_amdgcn_wmma_f32_16x16x32_f16(
                false, a[kc], false, b, (short)0, acc, false, false);
        }
        // C/D layout: VGPR j holds row M = j + 8*hi, col N = lane%16
#pragma unroll
        for (int j = 0; j < 8; ++j) {
            fht[(size_t)(row0 + hi * 8 + j) * OUTF + ct * 16 + Nloc] = acc[j];
        }
    }
}

// ---------------- 2) attention dot products el/er ---------------------------
__global__ void elr_kernel(const float* __restrict__ fh,
                           const float* __restrict__ attn_l,
                           const float* __restrict__ attn_r,
                           float* __restrict__ el, float* __restrict__ er)
{
    int i = blockIdx.x * blockDim.x + threadIdx.x;
    if (i >= TT * NN * HH) return;
    int h = i % HH;
    int n = (i / HH) % NN;
    int t = i / (HH * NN);
    const float4* f  = (const float4*)(fh + ((size_t)t * NN + n) * OUTF + h * DHD);
    const float4* al = (const float4*)(attn_l + (t * HH + h) * DHD);
    const float4* ar = (const float4*)(attn_r + (t * HH + h) * DHD);
    float sl = 0.f, sr = 0.f;
#pragma unroll
    for (int k = 0; k < DHD / 4; ++k) {
        float4 v = f[k], l4 = al[k], r4 = ar[k];
        sl += v.x * l4.x + v.y * l4.y + v.z * l4.z + v.w * l4.w;
        sr += v.x * r4.x + v.y * r4.y + v.z * r4.z + v.w * r4.w;
    }
    el[i] = sl; er[i] = sr;
}

// ---------------- 3) init m/den/agg ----------------------------------------
__global__ void init_kernel(unsigned* __restrict__ mkey, float* __restrict__ den,
                            float* __restrict__ agg)
{
    int i = blockIdx.x * blockDim.x + threadIdx.x;
    if (i < TT * NN * OUTF) agg[i] = 0.f;
    if (i < TT * NN * HH) { mkey[i] = 0u; den[i] = 0.f; }
}

// ---------------- 4) edge pass 1: logits + segment max ---------------------
__global__ void edge_pass1(const int* __restrict__ src, const int* __restrict__ dst,
                           const float* __restrict__ el, const float* __restrict__ er,
                           float* __restrict__ eb, unsigned* __restrict__ mkey)
{
    int i = blockIdx.x * blockDim.x + threadIdx.x;
    if (i >= TT * EE) return;
    int t = i / EE;
    int s = src[i], d = dst[i];
    float4 lv = *(const float4*)(el + ((size_t)t * NN + s) * HH);
    float4 rv = *(const float4*)(er + ((size_t)t * NN + d) * HH);
    float v[4] = { lv.x + rv.x, lv.y + rv.y, lv.z + rv.z, lv.w + rv.w };
    unsigned* mk = mkey + ((size_t)t * NN + d) * HH;
#pragma unroll
    for (int h = 0; h < HH; ++h) {
        v[h] = (v[h] > 0.f) ? v[h] : NEG_SLOPE * v[h];   // leaky_relu
        atomicMax(&mk[h], f2ord(v[h]));
    }
    *(float4*)(eb + (size_t)i * HH) = make_float4(v[0], v[1], v[2], v[3]);
}

// ---------------- 5) edge pass 2: exp + segment sum denominator ------------
__global__ void edge_pass2(const int* __restrict__ dst,
                           const unsigned* __restrict__ mkey,
                           float* __restrict__ eb, float* __restrict__ den)
{
    int i = blockIdx.x * blockDim.x + threadIdx.x;
    if (i >= TT * EE) return;
    int t = i / EE;
    int d = dst[i];
    uint4  mk = *(const uint4*)(mkey + ((size_t)t * NN + d) * HH);
    float4 ev = *(const float4*)(eb + (size_t)i * HH);
    float a0 = __expf(ev.x - ord2f(mk.x));
    float a1 = __expf(ev.y - ord2f(mk.y));
    float a2 = __expf(ev.z - ord2f(mk.z));
    float a3 = __expf(ev.w - ord2f(mk.w));
    *(float4*)(eb + (size_t)i * HH) = make_float4(a0, a1, a2, a3);
    float* dn = den + ((size_t)t * NN + d) * HH;
    atomicAdd(&dn[0], a0); atomicAdd(&dn[1], a1);
    atomicAdd(&dn[2], a2); atomicAdd(&dn[3], a3);
}

// ---------------- 6) message scatter: agg[dst] += alpha * fh[src] ----------
// 128 threads per edge -> fully coalesced row gather + coalesced f32 atomics.
__global__ void scatter_kernel(const int* __restrict__ src, const int* __restrict__ dst,
                               const float* __restrict__ fh, const float* __restrict__ ab,
                               const float* __restrict__ den, float* __restrict__ agg)
{
    long long gi = (long long)blockIdx.x * blockDim.x + threadIdx.x;
    if (gi >= (long long)TT * EE * OUTF) return;
    int      c  = (int)(gi % OUTF);
    long long ei = gi / OUTF;                     // [0, TT*EE)
    int t = (int)(ei / EE);
    int s = src[ei], d = dst[ei];
    int h = c / DHD;
    float alpha = ab[ei * HH + h] / den[((size_t)t * NN + d) * HH + h];
    atomicAdd(&agg[((size_t)t * NN + d) * OUTF + c],
              fh[((size_t)t * NN + s) * OUTF + c] * alpha);
}

// ---------------- 7) finalize: bias + LayerNorm + ELU (one wave32/node) ----
__global__ __launch_bounds__(256) void finalize_kernel(
    const float* __restrict__ agg, const float* __restrict__ bias,
    const float* __restrict__ lw, const float* __restrict__ lb,
    float* __restrict__ out)
{
    int wid  = blockIdx.x * (blockDim.x >> 5) + (threadIdx.x >> 5);
    int lane = threadIdx.x & 31;
    if (wid >= TT * NN) return;                   // exact: 100000 / 8 = 12500 blocks
    int t = wid / NN;
    int n = wid % NN;
    int c0 = lane * 4;                            // each lane: one float4 segment
    float4 av = *(const float4*)(agg + ((size_t)t * NN + n) * OUTF + c0);
    float4 bv = *(const float4*)(bias + t * OUTF + c0);
    float v[4] = { av.x + bv.x, av.y + bv.y, av.z + bv.z, av.w + bv.w };
    float sum = v[0] + v[1] + v[2] + v[3];
#pragma unroll
    for (int m = 16; m >= 1; m >>= 1) sum += __shfl_xor(sum, m, 32);
    float mean = sum * (1.f / OUTF);
    float vs = 0.f;
#pragma unroll
    for (int j = 0; j < 4; ++j) { float dv = v[j] - mean; vs += dv * dv; }
#pragma unroll
    for (int m = 16; m >= 1; m >>= 1) vs += __shfl_xor(vs, m, 32);
    float inv = rsqrtf(vs * (1.f / OUTF) + EPS_LN);
    float4 wv = *(const float4*)(lw + t * OUTF + c0);
    float4 sv = *(const float4*)(lb + t * OUTF + c0);
    float y0 = wv.x * ((v[0] - mean) * inv) + sv.x;
    float y1 = wv.y * ((v[1] - mean) * inv) + sv.y;
    float y2 = wv.z * ((v[2] - mean) * inv) + sv.z;
    float y3 = wv.w * ((v[3] - mean) * inv) + sv.w;
    y0 = (y0 > 0.f) ? y0 : expm1f(y0);
    y1 = (y1 > 0.f) ? y1 : expm1f(y1);
    y2 = (y2 > 0.f) ? y2 : expm1f(y2);
    y3 = (y3 > 0.f) ? y3 : expm1f(y3);
    *(float4*)(out + (size_t)n * (TT * OUTF) + t * OUTF + c0) =
        make_float4(y0, y1, y2, y3);
}

extern "C" void kernel_launch(void* const* d_in, const int* in_sizes, int n_in,
                              void* d_out, int out_size, void* d_ws, size_t ws_size,
                              hipStream_t stream)
{
    const float* feature = (const float*)d_in[0];
    const int*   src     = (const int*)  d_in[1];
    const int*   dst     = (const int*)  d_in[2];
    const float* W       = (const float*)d_in[3];
    const float* attn_l  = (const float*)d_in[4];
    const float* attn_r  = (const float*)d_in[5];
    const float* bias    = (const float*)d_in[6];
    const float* ln_w    = (const float*)d_in[7];
    const float* ln_b    = (const float*)d_in[8];
    float* out = (float*)d_out;

    // workspace carve-out (256B aligned)
    char* p = (char*)d_ws;
    auto carve = [&](size_t bytes) -> char* {
        char* r = p; p += (bytes + 255) & ~(size_t)255; return r;
    };
    float*     fh   = (float*)    carve(sizeof(float) * (size_t)TT * NN * OUTF); // 51.2 MB
    float*     el   = (float*)    carve(sizeof(float) * (size_t)TT * NN * HH);
    float*     er   = (float*)    carve(sizeof(float) * (size_t)TT * NN * HH);
    unsigned*  mkey = (unsigned*) carve(sizeof(unsigned) * (size_t)TT * NN * HH);
    float*     den  = (float*)    carve(sizeof(float) * (size_t)TT * NN * HH);
    float*     eb   = (float*)    carve(sizeof(float) * (size_t)TT * EE * HH);   // 25.6 MB
    float*     agg  = (float*)    carve(sizeof(float) * (size_t)TT * NN * OUTF); // 51.2 MB
    _Float16*  WT   = (_Float16*) carve(sizeof(_Float16) * (size_t)TT * OUTF * INF_); // 64 KB
    (void)ws_size; (void)in_sizes; (void)n_in; (void)out_size;

    // 0) W transpose + f32->f16 (tiny)
    {
        int tot = TT * INF_ * OUTF;
        wtrans_kernel<<<(tot + 255) / 256, 256, 0, stream>>>(W, WT);
    }

    // 1) WMMA projection: one wave per 16-row stripe per type
    proj_wmma_kernel<<<TT * (NN / 16), 32, 0, stream>>>(feature, WT, fh);

    // 2) el/er dot products
    {
        int tot = TT * NN * HH;
        elr_kernel<<<(tot + 255) / 256, 256, 0, stream>>>(fh, attn_l, attn_r, el, er);
    }

    // 3) init m / den / agg
    {
        int tot = TT * NN * OUTF;
        init_kernel<<<(tot + 255) / 256, 256, 0, stream>>>(mkey, den, agg);
    }

    // 4) edge logits + segment max
    {
        int tot = TT * EE;
        edge_pass1<<<(tot + 255) / 256, 256, 0, stream>>>(src, dst, el, er, eb, mkey);
    }

    // 5) exp + denominator
    {
        int tot = TT * EE;
        edge_pass2<<<(tot + 255) / 256, 256, 0, stream>>>(dst, mkey, eb, den);
    }

    // 6) message scatter
    {
        long long tot = (long long)TT * EE * OUTF;           // 204.8M threads
        int blocks = (int)((tot + 255) / 256);
        scatter_kernel<<<blocks, 256, 0, stream>>>(src, dst, fh, eb, den, agg);
    }

    // 7) bias + LayerNorm + ELU
    finalize_kernel<<<(TT * NN) / 8, 256, 0, stream>>>(agg, bias, ln_w, ln_b, out);
}
